// MultiHeadAttention_2173253452304
// MI455X (gfx1250) — matrix operand order repaired and checked by reading, not compiled, not run on hardware
//
#include <hip/hip_runtime.h>

#define B_  2
#define N_  256
#define M_  256
#define C_  256
#define H_  4
#define HC_ 64
#define SCALE_ 0.125f   // 1/sqrt(64)

typedef __attribute__((ext_vector_type(16))) _Float16 v16h;
typedef __attribute__((ext_vector_type(8)))  _Float16 v8h;
typedef __attribute__((ext_vector_type(8)))  float    v8f;
typedef __attribute__((ext_vector_type(4)))  float    v4f;

// ---------------------------------------------------------------------------
// Kernel 1: token projections  qp/kp/vp = tok @ W + b   (rows = b*256+t)
// ---------------------------------------------------------------------------
__global__ __launch_bounds__(256) void proj_kernel(
    const float* __restrict__ q_tok, const float* __restrict__ k_tok,
    const float* __restrict__ v_tok,
    const float* __restrict__ Wq, const float* __restrict__ bq,
    const float* __restrict__ Wk, const float* __restrict__ bk,
    const float* __restrict__ Wv, const float* __restrict__ bv,
    float* __restrict__ qp, float* __restrict__ kp, float* __restrict__ vp)
{
  const int row   = blockIdx.x;   // 0..511
  const int which = blockIdx.y;   // 0=q 1=k 2=v
  const float* src; const float* W; const float* bias; float* dst;
  if (which == 0)      { src = q_tok; W = Wq; bias = bq; dst = qp; }
  else if (which == 1) { src = k_tok; W = Wk; bias = bk; dst = kp; }
  else                 { src = v_tok; W = Wv; bias = bv; dst = vp; }

  __shared__ float srow[C_];
  const int t = threadIdx.x;
  srow[t] = src[row * C_ + t];
  __syncthreads();

  float acc = bias[t];
  #pragma unroll 4
  for (int c = 0; c < C_; ++c)
    acc += srow[c] * W[c * C_ + t];
  dst[row * C_ + t] = acc;
}

// ---------------------------------------------------------------------------
// Kernel 2: qtilde[b,n,h,c] = sum_{c'<64} qp[b,n,h*64+c'] * Wqk[c, h*64+c']
// ---------------------------------------------------------------------------
__global__ __launch_bounds__(256) void qtilde_kernel(
    const float* __restrict__ qp, const float* __restrict__ Wqk,
    float* __restrict__ qt)
{
  const int bn = blockIdx.x;   // 0..511
  const int h  = blockIdx.y;   // 0..3
  const int t  = threadIdx.x;  // output c

  __shared__ float qrow[HC_];
  if (t < HC_) qrow[t] = qp[bn * C_ + h * HC_ + t];
  __syncthreads();

  const float* wrow = Wqk + t * C_ + h * HC_;
  float acc = 0.f;
  #pragma unroll
  for (int c4 = 0; c4 < HC_; c4 += 4) {
    v4f w = *(const v4f*)(wrow + c4);
    acc += qrow[c4]     * w.x;
    acc += qrow[c4 + 1] * w.y;
    acc += qrow[c4 + 2] * w.z;
    acc += qrow[c4 + 3] * w.w;
  }
  qt[(bn * H_ + h) * C_ + t] = acc;
}

// ---------------------------------------------------------------------------
// Fragment helpers for v_wmma_f32_16x16x32_f16 (wave32)
//   A (16x32, f16): lane l -> row l%16; elems i     = K = k0+16*(l/16)+i
//   B (32x16, f16): lane l -> col l%16; elems i     = K = k0+16*(l/16)+i
// ---------------------------------------------------------------------------
__device__ __forceinline__ v16h load_a_frag(const _Float16* rowp, int half, int k0) {
  const int base = k0 + 16 * half;
  v8h lo = *(const v8h*)(rowp + base);
  v8h hi = *(const v8h*)(rowp + base + 8);
  v16h out;
  #pragma unroll
  for (int i = 0; i < 8; ++i) { out[i] = lo[i]; out[i + 8] = hi[i]; }
  return out;
}

// NT=true for read-once streaming tensors (qk/qv embeds): keep L2 for the
// heavily reused kp/vp/Wqv working set.
template <bool NT>
__device__ __forceinline__ v16h load_b_frag_f32(const float* __restrict__ p) {
  v4f x0, x1, x2, x3;
  if (NT) {
    x0 = __builtin_nontemporal_load((const v4f*)(p));
    x1 = __builtin_nontemporal_load((const v4f*)(p + 4));
    x2 = __builtin_nontemporal_load((const v4f*)(p + 8));
    x3 = __builtin_nontemporal_load((const v4f*)(p + 12));
  } else {
    x0 = *(const v4f*)(p);
    x1 = *(const v4f*)(p + 4);
    x2 = *(const v4f*)(p + 8);
    x3 = *(const v4f*)(p + 12);
  }
  v16h o;
  #pragma unroll
  for (int i = 0; i < 4; ++i) {
    o[i]      = (_Float16)x0[i];
    o[4 + i]  = (_Float16)x1[i];
    o[8 + i]  = (_Float16)x2[i];
    o[12 + i] = (_Float16)x3[i];
  }
  return o;
}

// ---------------------------------------------------------------------------
// Kernel 3: fused attention. One workgroup (256 thr = 8 waves) per (b,n).
// ---------------------------------------------------------------------------
__global__ __launch_bounds__(256) void fused_attn_kernel(
    const float* __restrict__ qp, const float* __restrict__ kp,
    const float* __restrict__ vp, const float* __restrict__ qt,
    const float* __restrict__ qk_embeds, const float* __restrict__ qv_embeds,
    const float* __restrict__ bqk, const float* __restrict__ bqv,
    const float* __restrict__ Wqv,
    float* __restrict__ hidden, float* __restrict__ attn_out)
{
  __shared__ __align__(16) _Float16 sQblk[16][C_];  // 8 KB  (A for q.k part)
  __shared__ __align__(16) _Float16 sQT[16][C_];    // 8 KB  (A for qtilde.E part)
  __shared__ float sQrow[C_];                       // 1 KB
  __shared__ float sScore[H_][M_];                  // 4 KB  (scores -> attn)
  __shared__ float sTmp[H_][64];                    // 1 KB  softmax partials
  __shared__ float sMax[H_], sSum[H_], sBias[H_];
  __shared__ float sU[4][H_][C_];                   // 16 KB per-group U partials
  __shared__ float sHV[4][C_];                      // 4 KB  per-group attn@v partials

  const int tid = threadIdx.x;
  const int bn  = blockIdx.x;
  const int b   = bn >> 8;
  const int n   = bn & 255;

  // ---- stage A-matrices (scale folded in) -------------------------------
  sQrow[tid] = qp[bn * C_ + tid];
  __syncthreads();
  #pragma unroll
  for (int i = tid; i < 16 * C_; i += 256) {
    const int r = i >> 8, c = i & 255;
    _Float16 qb = (_Float16)0.f, qtv = (_Float16)0.f;
    if (r < H_) {
      qtv = (_Float16)(qt[(bn * H_ + r) * C_ + c] * SCALE_);
      if ((c >> 6) == r) qb = (_Float16)(sQrow[c] * SCALE_);
    }
    sQblk[r][c] = qb;
    sQT[r][c]   = qtv;
  }
  if (tid < H_) {
    float acc = 0.f;
    for (int cc = 0; cc < HC_; ++cc)
      acc += sQrow[tid * HC_ + cc] * bqk[tid * HC_ + cc];
    sBias[tid] = acc * SCALE_;
  }
  __syncthreads();

  // ---- Phase A: scores via WMMA -----------------------------------------
  // S[h,m] = Qblk[h,:].kp[b,m,:] + QT[h,:].qk_embeds[b,n,m,:]  (scale folded)
  {
    const int wave = tid >> 5;
    const int lane = tid & 31;
    const int r    = lane & 15;
    const int half = lane >> 4;
    const float* kbase = kp + (size_t)b * (M_ * C_);
    const float* ebase = qk_embeds + (size_t)bn * (M_ * C_);

    #pragma unroll
    for (int t2 = 0; t2 < 2; ++t2) {
      const int m0 = (wave * 2 + t2) * 16;
      const float* brow_k = kbase + (m0 + r) * C_ + 16 * half;
      const float* brow_e = ebase + (size_t)(m0 + r) * C_ + 16 * half;
      v8f acc = {0.f, 0.f, 0.f, 0.f, 0.f, 0.f, 0.f, 0.f};
      #pragma unroll
      for (int k0 = 0; k0 < C_; k0 += 32) {
        v16h a  = load_a_frag(&sQblk[r][0], half, k0);
        v16h bb = load_b_frag_f32<false>(brow_k + k0);   // kp: L2-resident
        acc = __builtin_amdgcn_wmma_f32_16x16x32_f16(false, a, false, bb,
                                                     (short)0, acc, false, false);
      }
      #pragma unroll
      for (int k0 = 0; k0 < C_; k0 += 32) {
        v16h a  = load_a_frag(&sQT[r][0], half, k0);
        v16h bb = load_b_frag_f32<true>(brow_e + k0);    // embeds: stream NT
        acc = __builtin_amdgcn_wmma_f32_16x16x32_f16(false, a, false, bb,
                                                     (short)0, acc, false, false);
      }
      // D: lane l<16 holds D[M=j][N=l] in acc[j]; rows 0..3 = heads
      if (half == 0) {
        #pragma unroll
        for (int j = 0; j < H_; ++j)
          sScore[j][m0 + r] = acc[j] + sBias[j];
      }
    }
  }
  __syncthreads();

  // ---- Phase B: softmax over m per head; write attn ---------------------
  {
    const int h  = tid >> 6;
    const int j  = tid & 63;
    const int mb = j * 4;
    v4f s = *(const v4f*)&sScore[h][mb];
    sTmp[h][j] = fmaxf(fmaxf(s.x, s.y), fmaxf(s.z, s.w));
    __syncthreads();
    if (j == 0) {
      float mx = sTmp[h][0];
      for (int i = 1; i < 64; ++i) mx = fmaxf(mx, sTmp[h][i]);
      sMax[h] = mx;
    }
    __syncthreads();
    const float mx = sMax[h];
    const float e0 = __expf(s.x - mx), e1 = __expf(s.y - mx);
    const float e2 = __expf(s.z - mx), e3 = __expf(s.w - mx);
    sTmp[h][j] = e0 + e1 + e2 + e3;
    __syncthreads();
    if (j == 0) {
      float sm = 0.f;
      for (int i = 0; i < 64; ++i) sm += sTmp[h][i];
      sSum[h] = sm;
    }
    __syncthreads();
    const float inv = 1.f / sSum[h];
    v4f av = {e0 * inv, e1 * inv, e2 * inv, e3 * inv};
    *(v4f*)&sScore[h][mb] = av;
    float* ao = attn_out + (((size_t)(b * H_ + h)) * N_ + n) * M_ + mb;
    __builtin_nontemporal_store(av, (v4f*)ao);   // write-once output
  }
  __syncthreads();

  // ---- Phase C: U[h,:] = sum_m attn[h,m]*qv_embeds[m,:];
  //               HV[:]  = sum_m attn[c/64,m]*vp[b,m,:]   (rank-1, coalesced)
  {
    const int grp  = tid >> 6;            // m-range [grp*64, grp*64+64)
    const int c0   = (tid & 63) * 4;      // owns columns c0..c0+3
    const int hown = c0 >> 6;
    const float* Eqv   = qv_embeds + (size_t)bn * (M_ * C_);
    const float* vbase = vp + (size_t)b * (M_ * C_);
    float uacc[H_][4] = {};
    float hvacc[4] = {};
    const int mstart = grp * 64;
    for (int m = mstart; m < mstart + 64; ++m) {
      v4f e  = __builtin_nontemporal_load((const v4f*)(Eqv + (size_t)m * C_ + c0));
      v4f vv = *(const v4f*)(vbase + m * C_ + c0);
      const float a0 = sScore[0][m], a1 = sScore[1][m];
      const float a2 = sScore[2][m], a3 = sScore[3][m];
      const float ah = (hown == 0) ? a0 : (hown == 1) ? a1 : (hown == 2) ? a2 : a3;
      #pragma unroll
      for (int jj = 0; jj < 4; ++jj) {
        uacc[0][jj] += a0 * e[jj];
        uacc[1][jj] += a1 * e[jj];
        uacc[2][jj] += a2 * e[jj];
        uacc[3][jj] += a3 * e[jj];
        hvacc[jj]   += ah * vv[jj];
      }
    }
    #pragma unroll
    for (int jj = 0; jj < 4; ++jj) {
      sU[grp][0][c0 + jj] = uacc[0][jj];
      sU[grp][1][c0 + jj] = uacc[1][jj];
      sU[grp][2][c0 + jj] = uacc[2][jj];
      sU[grp][3][c0 + jj] = uacc[3][jj];
      sHV[grp][c0 + jj]   = hvacc[jj];
    }
  }
  __syncthreads();
  {
    const int c = tid;
    #pragma unroll
    for (int h = 0; h < H_; ++h)
      sU[0][h][c] = sU[0][h][c] + sU[1][h][c] + sU[2][h][c] + sU[3][h][c];
    sHV[0][c] = sHV[0][c] + sHV[1][c] + sHV[2][c] + sHV[3][c];
  }
  __syncthreads();

  // ---- final: hidden[c] = HV[c] + bqv[c] + sum_c' U[h,c']*Wqv[c',c] -----
  {
    const int cout = tid;
    const int h    = cout >> 6;
    float acc = sHV[0][cout] + bqv[cout];
    #pragma unroll 4
    for (int c = 0; c < C_; ++c)
      acc += sU[0][h][c] * Wqv[c * C_ + cout];
    __builtin_nontemporal_store(acc, &hidden[(size_t)bn * C_ + cout]);
  }
}

// ---------------------------------------------------------------------------
extern "C" void kernel_launch(void* const* d_in, const int* in_sizes, int n_in,
                              void* d_out, int out_size, void* d_ws, size_t ws_size,
                              hipStream_t stream) {
  const float* q_tok = (const float*)d_in[0];
  const float* k_tok = (const float*)d_in[1];
  const float* v_tok = (const float*)d_in[2];
  const float* qk_e  = (const float*)d_in[3];
  const float* qv_e  = (const float*)d_in[4];
  const float* Wq  = (const float*)d_in[5];
  const float* bq  = (const float*)d_in[6];
  const float* Wk  = (const float*)d_in[7];
  const float* bk  = (const float*)d_in[8];
  const float* Wv  = (const float*)d_in[9];
  const float* bv  = (const float*)d_in[10];
  const float* Wqk = (const float*)d_in[11];
  const float* bqk = (const float*)d_in[12];
  const float* Wqv = (const float*)d_in[13];
  const float* bqv = (const float*)d_in[14];

  float* ws = (float*)d_ws;
  float* qp = ws;                 // B*N*C   = 131072
  float* kp = ws + 131072;        // B*M*C   = 131072
  float* vp = ws + 262144;        // B*M*C   = 131072
  float* qt = ws + 393216;        // B*N*H*C = 524288
  float* hidden = (float*)d_out;                        // B*N*C
  float* attn   = (float*)d_out + (size_t)B_ * N_ * C_; // B*H*N*M

  proj_kernel<<<dim3(B_ * N_, 3), 256, 0, stream>>>(
      q_tok, k_tok, v_tok, Wq, bq, Wk, bk, Wv, bv, qp, kp, vp);
  qtilde_kernel<<<dim3(B_ * N_, H_), 256, 0, stream>>>(qp, Wqk, qt);
  fused_attn_kernel<<<B_ * N_, 256, 0, stream>>>(
      qp, kp, vp, qt, qk_e, qv_e, bqk, bqv, Wqv, hidden, attn);
}
